// temporal_model_67061619359899
// MI455X (gfx1250) — matrix operand order repaired and checked by reading, not compiled, not run on hardware
//
#include <hip/hip_runtime.h>
#include <hip/hip_bf16.h>

// Problem constants (from reference)
#define B_  128
#define V_  24
#define T_  30
#define P_  15
#define F_  4
#define H_  256
#define NROWS (B_*V_)          // 3072 independent rows
#define G4_  (4*H_)            // 1024 gate columns
#define NT_  (G4_/16)          // 64 N-tiles of the gate matrix
#define KT_  (H_/32)           // 8 K-tiles (bf16 wmma K=32)
#define TILE_DW 256            // dwords per pre-swizzled 32x16 bf16 B tile
#define WTILE_DW (KT_*NT_*TILE_DW)   // 131072 dwords (512 KB) per weight matrix

typedef __attribute__((ext_vector_type(16))) __bf16 v16bf;
typedef __attribute__((ext_vector_type(8)))  float  v8f;

union FragAB { uint4 q[2]; v16bf v; };

__device__ __forceinline__ unsigned short f32_bf16(float f) {
  unsigned int u = __float_as_uint(f);
  u += 0x7FFFu + ((u >> 16) & 1u);   // round-to-nearest-even
  return (unsigned short)(u >> 16);
}
// Branch-free activations on v_exp_f32 (no ocml range-check divergence).
__device__ __forceinline__ float sigf(float x)  { return 1.0f / (1.0f + __expf(-x)); }
__device__ __forceinline__ float tanhfast(float x) {
  // 1 - 2/(e^{2x}+1): exact limits (+1/-1) at +/-inf, no inf/inf NaN.
  return 1.0f - 2.0f / (__expf(2.0f * x) + 1.0f);
}

// ---------------------------------------------------------------------------
// Prep: swizzle W_hh (4H x H, row-major fp32) into WMMA B-fragment tiles.
// B = W^T (K=256 x N=1024). Tile (kt,nt): lane L holds 8 dwords (16 bf16):
//   N = nt*16 + (L&15),  K = kt*32 + (v>>2)*16 + (L>>4)*8 + 2(v&3) + h   (v=dword, h=half)
// Stored so lane L reads dwords [ (tile*32+L)*8 .. +8 ) -> two b128 loads.
// ---------------------------------------------------------------------------
__global__ void prep_weights(const float* __restrict__ Wenc,
                             const float* __restrict__ Wdec,
                             unsigned int* __restrict__ tiles) {
  int idx = blockIdx.x * blockDim.x + threadIdx.x;  // dword index
  if (idx >= 2 * WTILE_DW) return;
  const float* W = (idx < WTILE_DW) ? Wenc : Wdec;
  int d    = idx & (WTILE_DW - 1);
  int v    = d & 7;
  int lane = (d >> 3) & 31;
  int tile = d >> 8;            // kt*NT_ + nt
  int nt   = tile & (NT_ - 1);
  int kt   = tile >> 6;
  int n    = nt * 16 + (lane & 15);
  int k0   = kt * 32 + (v >> 2) * 16 + (lane >> 4) * 8 + 2 * (v & 3);
  unsigned int lo = f32_bf16(W[n * H_ + k0]);
  unsigned int hi = f32_bf16(W[n * H_ + k0 + 1]);
  tiles[idx] = lo | (hi << 16);
}

// ---------------------------------------------------------------------------
// Main persistent kernel: one block = 16 rows, 8 wave32s.
// Wave w owns hidden columns [32w, 32w+32) of all 4 gates (i,f,g,o), so
// c-state and LSTM pointwise math stay in registers across all 45 steps.
// B-tiles deliberately stream from L2 each step: the kt loop is kept rolled
// so the loads cannot be hoisted across the recurrence and spilled; the
// laundered *offset* keeps global-addrspace provenance (global_load_b128,
// not flat) while defeating cross-step load CSE.
// ---------------------------------------------------------------------------
__global__ __launch_bounds__(256) void temporal_main(
    const float* __restrict__ sequence,        // (B,V,T,F)
    const unsigned char* __restrict__ seq_mask,// (B,V,T) bool
    const float* __restrict__ W_ih_enc,        // (4H,F)
    const float* __restrict__ b_ih_enc,
    const float* __restrict__ b_hh_enc,
    const float* __restrict__ W_ih_dec,
    const float* __restrict__ b_ih_dec,
    const float* __restrict__ b_hh_dec,
    const float* __restrict__ W_out,           // (F,H)
    const float* __restrict__ b_out,           // (F)
    const unsigned int* __restrict__ tiles_enc,
    const unsigned int* __restrict__ tiles_dec,
    float* __restrict__ enc_ws,                // (NROWS,T,H) fp32 workspace
    float* __restrict__ out)                   // output(184320) ++ align(1382400)
{
  const int tid  = threadIdx.x;
  const int lane = tid & 31;
  const int w    = tid >> 5;
  const int base = blockIdx.x * 16;            // first global row of this tile

  __shared__ __align__(16) unsigned short h_bf[2][16 * H_]; // bf16 h (A-operand), dbl-buf
  __shared__ __align__(16) float h_f[16 * H_];              // fp32 h (attention/pred)
  __shared__ float x_sm[16 * F_];
  __shared__ float m_sm[16];
  __shared__ float msk_sm[16 * T_];
  __shared__ float sc_sm[16 * T_];
  __shared__ float prev_sm[16 * F_];
  __shared__ float opm_sm[16];
  __shared__ __align__(16) float wout_sm[F_ * H_];
  __shared__ float bout_sm[F_];

  // ---- one-time init ----
  for (int i = tid; i < 16 * H_; i += 256) { h_bf[0][i] = 0; h_f[i] = 0.0f; }
  for (int i = tid; i < F_ * H_; i += 256) wout_sm[i] = W_out[i];
  if (tid < F_) bout_sm[tid] = b_out[tid];
  for (int i = tid; i < 16 * T_; i += 256) {
    int r = i / T_, t = i % T_;
    msk_sm[i] = seq_mask[(size_t)(base + r) * T_ + t] ? 1.0f : 0.0f;
  }

  // per-wave column ownership + encoder input-projection weights (invariant)
  int   ncol[8];
  float wih[8][4], bias[8];
#pragma unroll
  for (int u = 0; u < 8; ++u) {
    int g = u >> 1, j = u & 1;
    int n = g * H_ + w * 32 + j * 16 + (lane & 15);
    ncol[u] = n;
    bias[u] = b_ih_enc[n] + b_hh_enc[n];
#pragma unroll
    for (int f = 0; f < 4; ++f) wih[u][f] = W_ih_enc[n * 4 + f];
  }
  float creg[2][8];
#pragma unroll
  for (int j = 0; j < 2; ++j)
#pragma unroll
    for (int v = 0; v < 8; ++v) creg[j][v] = 0.0f;

  const int arow = lane & 15;
  const int koff = (lane >> 4) * 8;

  __syncthreads();

  // =========================== ENCODER ===========================
  int cur = 0;
  for (int t = 0; t < T_; ++t) {
    if (tid < 64) {
      int r = tid >> 2, f = tid & 3;
      x_sm[tid] = sequence[(((size_t)(base + r)) * T_ + t) * F_ + f];
    }
    if (tid < 16) m_sm[tid] = msk_sm[tid * T_ + t];
    __syncthreads();

    float xr[8][4];
#pragma unroll
    for (int v = 0; v < 8; ++v) {
      int r = v + (lane >> 4) * 8;
#pragma unroll
      for (int f = 0; f < 4; ++f) xr[v][f] = x_sm[r * 4 + f];
    }
    v8f acc[8];
#pragma unroll
    for (int u = 0; u < 8; ++u)
#pragma unroll
      for (int v = 0; v < 8; ++v)
        acc[u][v] = bias[u] + xr[v][0] * wih[u][0] + xr[v][1] * wih[u][1]
                            + xr[v][2] * wih[u][2] + xr[v][3] * wih[u][3];

    // ---- gates += h @ Whh^T : rolled K loop, B streams from L2 ----
    {
      size_t ztile = 0;
      asm volatile("" : "+s"(ztile));          // opaque 0: defeats cross-step CSE,
      const unsigned int* tbase = tiles_enc + ztile; // keeps global provenance
      const unsigned short* hb = h_bf[cur];
#pragma clang loop unroll(disable)
      for (int kt = 0; kt < KT_; ++kt) {
        FragAB af;
        af.q[0] = *(const uint4*)&hb[arow * H_ + kt * 32 + koff];
        af.q[1] = *(const uint4*)&hb[arow * H_ + kt * 32 + 16 + koff];
#pragma unroll
        for (int u = 0; u < 8; ++u) {
          int g = u >> 1, j = u & 1;
          int nt = g * 16 + w * 2 + j;
          const uint4* bp = (const uint4*)&tbase[((kt * NT_ + nt) * 32 + lane) * 8];
          FragAB bf; bf.q[0] = bp[0]; bf.q[1] = bp[1];
          acc[u] = __builtin_amdgcn_wmma_f32_16x16x32_bf16(
              false, af.v, false, bf.v, (short)0, acc[u], false, false);
        }
      }
    }

    // pointwise LSTM cell entirely in registers; c is never spilled
    int nxt = cur ^ 1;
#pragma unroll
    for (int j = 0; j < 2; ++j) {
      int nh = w * 32 + j * 16 + (lane & 15);
#pragma unroll
      for (int v = 0; v < 8; ++v) {
        int r = v + (lane >> 4) * 8;
        float ig = sigf(acc[0 + j][v]);
        float fg = sigf(acc[2 + j][v]);
        float gg = tanhfast(acc[4 + j][v]);
        float og = sigf(acc[6 + j][v]);
        float cn = fg * creg[j][v] + ig * gg;
        float hv = og * tanhfast(cn) * m_sm[r];
        creg[j][v] = cn;
        h_bf[nxt][r * H_ + nh] = f32_bf16(hv);
        if (t == T_ - 1) h_f[r * H_ + nh] = hv;  // seed decoder attention
        enc_ws[(((size_t)(base + r)) * T_ + t) * H_ + nh] = hv;
      }
    }
    __syncthreads();
    cur = nxt;
  }

  // =========================== DECODER ===========================
  if (tid < 16) opm_sm[tid] = msk_sm[tid * T_ + (T_ - 1)];
  if (tid < 64) {
    int r = tid >> 2, f = tid & 3;
    prev_sm[tid] = sequence[(((size_t)(base + r)) * T_ + (T_ - 1)) * F_ + f];
  }
  float wihd[8][4], biasd[8];
#pragma unroll
  for (int u = 0; u < 8; ++u) {
    int n = ncol[u];
    biasd[u] = b_ih_dec[n] + b_hh_dec[n];
#pragma unroll
    for (int f = 0; f < 4; ++f) wihd[u][f] = W_ih_dec[n * 4 + f];
  }
  __syncthreads();

  const size_t out_align = (size_t)NROWS * P_ * F_;   // 184320
  for (int p = 0; p < P_; ++p) {
    // ---- attention scores: encoded . h over H (fp32, L2 resident) ----
    for (int idx = tid; idx < 16 * T_; idx += 256) {
      int r = idx / T_, t = idx % T_;
      const float4* ep = (const float4*)&enc_ws[(((size_t)(base + r)) * T_ + t) * H_];
      const float4* hp = (const float4*)&h_f[r * H_];
      float s = 0.0f;
      for (int k = 0; k < H_ / 4; ++k) {
        float4 a = ep[k], b = hp[k];
        s += a.x * b.x + a.y * b.y + a.z * b.z + a.w * b.w;
      }
      sc_sm[idx] = (msk_sm[idx] > 0.0f) ? s : -1.0e9f;
    }
    __syncthreads();
    // ---- softmax per row + alignments output ----
    if (tid < 16) {
      int r = tid;
      float mx = -3.4e38f;
      for (int t = 0; t < T_; ++t) mx = fmaxf(mx, sc_sm[r * T_ + t]);
      float sum = 0.0f;
      for (int t = 0; t < T_; ++t) { float e = __expf(sc_sm[r * T_ + t] - mx); sc_sm[r * T_ + t] = e; sum += e; }
      float inv = 1.0f / sum;
      for (int t = 0; t < T_; ++t) {
        float a = sc_sm[r * T_ + t] * inv;
        sc_sm[r * T_ + t] = a;
        out[out_align + ((size_t)p * NROWS + base + r) * T_ + t] = a;
      }
    }
    __syncthreads();
    // ---- ctx = alpha . encoded, written as bf16 A-operand ----
    for (int idx = tid; idx < 16 * H_; idx += 256) {
      int r = idx >> 8, k = idx & (H_ - 1);
      const float* ep = &enc_ws[((size_t)(base + r)) * T_ * H_ + k];
      float c = 0.0f;
      for (int t = 0; t < T_; ++t) c += sc_sm[r * T_ + t] * ep[t * H_];
      h_bf[0][idx] = f32_bf16(c);
    }
    __syncthreads();
    // ---- decoder LSTM gates: prev@Wih^T (VALU) + ctx@Whh^T (WMMA) ----
    float xr[8][4];
#pragma unroll
    for (int v = 0; v < 8; ++v) {
      int r = v + (lane >> 4) * 8;
#pragma unroll
      for (int f = 0; f < 4; ++f) xr[v][f] = prev_sm[r * 4 + f];
    }
    v8f acc[8];
#pragma unroll
    for (int u = 0; u < 8; ++u)
#pragma unroll
      for (int v = 0; v < 8; ++v)
        acc[u][v] = biasd[u] + xr[v][0] * wihd[u][0] + xr[v][1] * wihd[u][1]
                             + xr[v][2] * wihd[u][2] + xr[v][3] * wihd[u][3];
    {
      size_t ztile = 0;
      asm volatile("" : "+s"(ztile));
      const unsigned int* tbase = tiles_dec + ztile;
      const unsigned short* hb = h_bf[0];
#pragma clang loop unroll(disable)
      for (int kt = 0; kt < KT_; ++kt) {
        FragAB af;
        af.q[0] = *(const uint4*)&hb[arow * H_ + kt * 32 + koff];
        af.q[1] = *(const uint4*)&hb[arow * H_ + kt * 32 + 16 + koff];
#pragma unroll
        for (int u = 0; u < 8; ++u) {
          int g = u >> 1, j = u & 1;
          int nt = g * 16 + w * 2 + j;
          const uint4* bp = (const uint4*)&tbase[((kt * NT_ + nt) * 32 + lane) * 8];
          FragAB bf; bf.q[0] = bp[0]; bf.q[1] = bp[1];
          acc[u] = __builtin_amdgcn_wmma_f32_16x16x32_bf16(
              false, af.v, false, bf.v, (short)0, acc[u], false, false);
        }
      }
    }
#pragma unroll
    for (int j = 0; j < 2; ++j) {
      int nh = w * 32 + j * 16 + (lane & 15);
#pragma unroll
      for (int v = 0; v < 8; ++v) {
        int r = v + (lane >> 4) * 8;
        float ig = sigf(acc[0 + j][v]);
        float fg = sigf(acc[2 + j][v]);
        float gg = tanhfast(acc[4 + j][v]);
        float og = sigf(acc[6 + j][v]);
        float cn = fg * creg[j][v] + ig * gg;      // c2 is NOT masked
        float hv = og * tanhfast(cn) * opm_sm[r];  // h2 masked by opm
        creg[j][v] = cn;
        h_f[r * H_ + nh] = hv;
      }
    }
    __syncthreads();
    // ---- pred = relu(h2 @ W_out^T + b_out), masked; feeds next step ----
    if (tid < 64) {
      int r = tid >> 2, f = tid & 3;
      const float4* hp = (const float4*)&h_f[r * H_];
      const float4* wp = (const float4*)&wout_sm[f * H_];
      float s = bout_sm[f];
      for (int k = 0; k < H_ / 4; ++k) {
        float4 a = hp[k], b = wp[k];
        s += a.x * b.x + a.y * b.y + a.z * b.z + a.w * b.w;
      }
      s = fmaxf(s, 0.0f) * opm_sm[r];
      prev_sm[tid] = s;
      out[(((size_t)(base + r)) * P_ + p) * F_ + f] = s;
    }
    __syncthreads();
  }
}

// ---------------------------------------------------------------------------
extern "C" void kernel_launch(void* const* d_in, const int* in_sizes, int n_in,
                              void* d_out, int out_size, void* d_ws, size_t ws_size,
                              hipStream_t stream) {
  (void)in_sizes; (void)n_in; (void)out_size; (void)ws_size;
  const float*         sequence  = (const float*)d_in[0];
  // d_in[1..3]: dist/bearing/heading matrices — unused by the reference graph
  const unsigned char* seq_mask  = (const unsigned char*)d_in[4];
  // d_in[5]: op_mask — unused (reference derives opm from seq_mask[:,:,-1])
  const float*         W_ih_enc  = (const float*)d_in[6];
  const float*         W_hh_enc  = (const float*)d_in[7];
  const float*         b_ih_enc  = (const float*)d_in[8];
  const float*         b_hh_enc  = (const float*)d_in[9];
  const float*         W_ih_dec  = (const float*)d_in[10];
  const float*         W_hh_dec  = (const float*)d_in[11];
  const float*         b_ih_dec  = (const float*)d_in[12];
  const float*         b_hh_dec  = (const float*)d_in[13];
  const float*         W_out     = (const float*)d_in[14];
  const float*         b_out     = (const float*)d_in[15];

  // Workspace: [ enc tiles 512KB | dec tiles 512KB | encoded fp32 94.4MB ]
  unsigned int* tiles_enc = (unsigned int*)d_ws;
  unsigned int* tiles_dec = tiles_enc + WTILE_DW;
  float*        enc_ws    = (float*)(tiles_enc + 2 * WTILE_DW);

  prep_weights<<<(2 * WTILE_DW + 255) / 256, 256, 0, stream>>>(W_hh_enc, W_hh_dec, tiles_enc);
  temporal_main<<<NROWS / 16, 256, 0, stream>>>(
      sequence, seq_mask, W_ih_enc, b_ih_enc, b_hh_enc,
      W_ih_dec, b_ih_dec, b_hh_dec, W_out, b_out,
      tiles_enc, tiles_dec, enc_ws, (float*)d_out);
}
